// ParticleNet_47562467836729
// MI455X (gfx1250) — compile-verified, matrix operand order checked by reading, not run on hardware
//
#include <hip/hip_runtime.h>
#include <math.h>

typedef __attribute__((ext_vector_type(16))) _Float16 v16h;
typedef __attribute__((ext_vector_type(8)))  _Float16 v8h;
typedef __attribute__((ext_vector_type(8)))  float    v8f;

#define G_GRP 64
#define N_PTS 128
#define K_NN  16
#define EPSV  1e-5f

// ---------------------------------------------------------------------------
// Utility kernels
// ---------------------------------------------------------------------------
__global__ void pn_zero(float* __restrict__ p, int n) {
    for (int i = blockIdx.x * blockDim.x + threadIdx.x; i < n;
         i += gridDim.x * blockDim.x) p[i] = 0.0f;
}

// W (Kd x Nc) f32 row-major -> WT (Nc x Kp) f16 row-major, K-padded with zeros
__global__ void pn_cvt_wT(const float* __restrict__ W, _Float16* __restrict__ WT,
                          int Kd, int Nc, int Kp) {
    size_t tot = (size_t)Nc * Kp;
    for (size_t i = (size_t)blockIdx.x * blockDim.x + threadIdx.x; i < tot;
         i += (size_t)gridDim.x * blockDim.x) {
        int n = (int)(i / Kp), k = (int)(i % Kp);
        WT[i] = (k < Kd) ? (_Float16)W[(size_t)k * Nc + n] : (_Float16)0.0f;
    }
}

// per-column sum / sumsq via atomics
__global__ void pn_colstats(const float* __restrict__ X, int M, int C,
                            float* __restrict__ sum, float* __restrict__ sumsq) {
    size_t tot = (size_t)M * C;
    for (size_t i = (size_t)blockIdx.x * blockDim.x + threadIdx.x; i < tot;
         i += (size_t)gridDim.x * blockDim.x) {
        int c = (int)(i % C);
        float v = X[i];
        atomicAdd(sum + c, v);
        atomicAdd(sumsq + c, v * v);
    }
}

// normalize input with batch stats -> f32 (stride C) + f16 (stride ldh, pre-zeroed)
__global__ void pn_in_norm(const float* __restrict__ X, const float* __restrict__ sum,
                           const float* __restrict__ sumsq, const float* __restrict__ g,
                           const float* __restrict__ b, float* __restrict__ F32,
                           _Float16* __restrict__ F16, int M, int C, int ldh) {
    size_t tot = (size_t)M * C;
    float invM = 1.0f / (float)M;
    for (size_t i = (size_t)blockIdx.x * blockDim.x + threadIdx.x; i < tot;
         i += (size_t)gridDim.x * blockDim.x) {
        int c = (int)(i % C);
        int m = (int)(i / C);
        float mean = sum[c] * invM;
        float var  = sumsq[c] * invM - mean * mean;
        float y = (X[i] - mean) * rsqrtf(var + EPSV) * g[c] + b[c];
        F32[i] = y;
        F16[(size_t)m * ldh + c] = (_Float16)y;
    }
}

// ---------------------------------------------------------------------------
// kNN: one workgroup per group; points staged in dynamic LDS
// ---------------------------------------------------------------------------
__global__ void pn_knn(const float* __restrict__ P, int stride, int D,
                       int* __restrict__ nbr) {
    extern __shared__ float sp[];                   // N_PTS * D floats
    int g = blockIdx.x;
    int t = threadIdx.x;                            // 0..127
    for (int i = t; i < N_PTS * D; i += blockDim.x)
        sp[i] = P[(size_t)(g * N_PTS + i / D) * stride + (i % D)];
    __syncthreads();

    float bd[K_NN]; int bi[K_NN];
#pragma unroll
    for (int k = 0; k < K_NN; ++k) { bd[k] = 3.4e38f; bi[k] = 0; }

    const float* pn = sp + t * D;
    for (int m = 0; m < N_PTS; ++m) {
        if (m == t) continue;                       // diagonal excluded (+1e9 in ref)
        const float* pm = sp + m * D;
        float d = 0.0f;
        for (int c = 0; c < D; ++c) { float df = pn[c] - pm[c]; d += df * df; }
        if (d < bd[K_NN - 1]) {
            int pos = K_NN - 1;
            while (pos > 0 && bd[pos - 1] > d) {
                bd[pos] = bd[pos - 1]; bi[pos] = bi[pos - 1]; --pos;
            }
            bd[pos] = d; bi[pos] = m;
        }
    }
#pragma unroll
    for (int k = 0; k < K_NN; ++k)
        nbr[(size_t)(g * N_PTS + t) * K_NN + k] = g * N_PTS + bi[k];
}

// ---------------------------------------------------------------------------
// Branch-free WMMA GEMM: Z(grid.x*128 x Nc) = A * WT^T
// A: f16, row stride Kp (multiple of 32, zero-padded); WT: Nc x Kp f16.
// Per K-step each lane does two b128 loads per operand (ISA f16 layout:
// elements 0-7 = halves [k0+khalf .. +7], 8-15 = [k0+16+khalf .. +7]).
// ---------------------------------------------------------------------------
__global__ __launch_bounds__(256)
void pn_gemm(const _Float16* __restrict__ A, const _Float16* __restrict__ BT,
             float* __restrict__ Z, int Nc, int Kp,
             float* __restrict__ sum, float* __restrict__ sumsq) {
    const int lane  = threadIdx.x & 31;
    const int wave  = threadIdx.x >> 5;
    const int m0    = blockIdx.x * 128 + wave * 16;
    const int n0    = blockIdx.y * 16;
    const int l15   = lane & 15;
    const int khalf = (lane >> 4) << 3;             // 0 or 8
    const _Float16* ap = A  + (size_t)(m0 + l15) * Kp + khalf;
    const _Float16* bp = BT + (size_t)(n0 + l15) * Kp + khalf;

    v8f acc = {};
    for (int k0 = 0; k0 < Kp; k0 += 32) {
        v16h a, b;
        ((v8h*)&a)[0] = *(const v8h*)(ap + k0);
        ((v8h*)&a)[1] = *(const v8h*)(ap + k0 + 16);
        ((v8h*)&b)[0] = *(const v8h*)(bp + k0);
        ((v8h*)&b)[1] = *(const v8h*)(bp + k0 + 16);
        acc = __builtin_amdgcn_wmma_f32_16x16x32_f16(false, a, false, b,
                                                     (short)0, acc, false, false);
    }

    float ls = 0.0f, ls2 = 0.0f;
#pragma unroll
    for (int r = 0; r < 8; ++r) {
        float v = acc[r];
        Z[(size_t)(m0 + r + khalf) * Nc + n0 + l15] = v;
        ls += v; ls2 += v * v;
    }
    if (sum != nullptr) {
        atomicAdd(sum   + n0 + l15, ls);
        atomicAdd(sumsq + n0 + l15, ls2);
    }
}

// ---------------------------------------------------------------------------
// Edge-feature WMMA GEMM (fused gather, branch-free A synthesis):
// row r -> (gn = r/16); col c < in_c -> fts[gn][c]; else fts[nbr[r]][c-in_c]
// - fts[gn][c-in_c]; cols >= 2*in_c are zero padding. Index clamping +
// cndmask selects keep all loads unconditional (no exec branching).
// ---------------------------------------------------------------------------
__device__ __forceinline__ float pn_edge_elem(const float* __restrict__ fi,
                                              const float* __restrict__ fj,
                                              int c, int in_c, int Kd) {
    int cc = (c < in_c) ? c : (c - in_c);
    int cs = (c < Kd) ? cc : 0;
    float vi = fi[cs];
    float vj = fj[cs];
    float v  = (c < in_c) ? vi : (vj - vi);
    return (c < Kd) ? v : 0.0f;
}

__global__ __launch_bounds__(256)
void pn_gemm_edge(const float* __restrict__ fts, const int* __restrict__ nbr,
                  int in_c, const _Float16* __restrict__ BT,
                  float* __restrict__ Z, int Nc, int Kp,
                  float* __restrict__ sum, float* __restrict__ sumsq) {
    const int Kd    = 2 * in_c;
    const int lane  = threadIdx.x & 31;
    const int wave  = threadIdx.x >> 5;
    const int m0    = blockIdx.x * 128 + wave * 16;
    const int n0    = blockIdx.y * 16;
    const int l15   = lane & 15;
    const int khalf = (lane >> 4) << 3;
    const int mrow  = m0 + l15;

    const int nb = nbr[mrow];
    const float* fi = fts + (size_t)(mrow >> 4) * in_c;  // uniform across wave tile
    const float* fj = fts + (size_t)nb * in_c;
    const _Float16* bp = BT + (size_t)(n0 + l15) * Kp + khalf;

    v8f acc = {};
    for (int k0 = 0; k0 < Kp; k0 += 32) {
        v16h a, b;
        ((v8h*)&b)[0] = *(const v8h*)(bp + k0);
        ((v8h*)&b)[1] = *(const v8h*)(bp + k0 + 16);
#pragma unroll
        for (int j = 0; j < 8; ++j) {
            int kk = k0 + ((j & 4) << 2) + khalf + ((j & 3) << 1);
            a[2 * j]     = (_Float16)pn_edge_elem(fi, fj, kk,     in_c, Kd);
            a[2 * j + 1] = (_Float16)pn_edge_elem(fi, fj, kk + 1, in_c, Kd);
        }
        acc = __builtin_amdgcn_wmma_f32_16x16x32_f16(false, a, false, b,
                                                     (short)0, acc, false, false);
    }

    float ls = 0.0f, ls2 = 0.0f;
#pragma unroll
    for (int r = 0; r < 8; ++r) {
        float v = acc[r];
        Z[(size_t)(m0 + r + khalf) * Nc + n0 + l15] = v;
        ls += v; ls2 += v * v;
    }
    if (sum != nullptr) {
        atomicAdd(sum   + n0 + l15, ls);
        atomicAdd(sumsq + n0 + l15, ls2);
    }
}

// ---------------------------------------------------------------------------
// BN (batch stats) + ReLU ; writes f16 operand for next GEMM
// ---------------------------------------------------------------------------
__global__ void pn_bn_relu(const float* __restrict__ Z, _Float16* __restrict__ H,
                           const float* __restrict__ sum, const float* __restrict__ sumsq,
                           const float* __restrict__ g, const float* __restrict__ b,
                           int M, int C) {
    size_t tot = (size_t)M * C;
    float invM = 1.0f / (float)M;
    for (size_t i = (size_t)blockIdx.x * blockDim.x + threadIdx.x; i < tot;
         i += (size_t)gridDim.x * blockDim.x) {
        int c = (int)(i % C);
        float mean = sum[c] * invM;
        float var  = sumsq[c] * invM - mean * mean;
        float y = (Z[i] - mean) * rsqrtf(var + EPSV) * g[c] + b[c];
        H[i] = (_Float16)fmaxf(y, 0.0f);
    }
}

// max over K neighbors + BN'd skip + ReLU -> new fts (f32 + f16)
__global__ void pn_combine(const _Float16* __restrict__ H3, const float* __restrict__ Zs,
                           const float* __restrict__ sumS, const float* __restrict__ sumsqS,
                           const float* __restrict__ gs, const float* __restrict__ bs,
                           float* __restrict__ F32, _Float16* __restrict__ F16,
                           int Mgn, int C) {
    size_t tot = (size_t)Mgn * C;
    float invM = 1.0f / (float)Mgn;
    for (size_t i = (size_t)blockIdx.x * blockDim.x + threadIdx.x; i < tot;
         i += (size_t)gridDim.x * blockDim.x) {
        int c  = (int)(i % C);
        int gn = (int)(i / C);
        float mx = -3.4e38f;
#pragma unroll
        for (int k = 0; k < K_NN; ++k)
            mx = fmaxf(mx, (float)H3[((size_t)gn * K_NN + k) * C + c]);
        float mean = sumS[c] * invM;
        float var  = sumsqS[c] * invM - mean * mean;
        float skip = (Zs[i] - mean) * rsqrtf(var + EPSV) * gs[c] + bs[c];
        float y = fmaxf(mx + skip, 0.0f);
        F32[i] = y;
        F16[i] = (_Float16)y;
    }
}

// mean over N points -> pooled (f16 operand for FC GEMM; rows 64..127 pre-zeroed)
__global__ void pn_pool(const float* __restrict__ F32, _Float16* __restrict__ pooled, int C) {
    int tot = G_GRP * C;
    for (int i = blockIdx.x * blockDim.x + threadIdx.x; i < tot;
         i += gridDim.x * blockDim.x) {
        int g = i / C, c = i % C;
        float s = 0.0f;
        for (int n = 0; n < N_PTS; ++n)
            s += F32[(size_t)(g * N_PTS + n) * C + c];
        pooled[i] = (_Float16)(s * (1.0f / (float)N_PTS));
    }
}

__global__ void pn_bias_relu(float* __restrict__ Z, const float* __restrict__ b,
                             int M, int C) {
    size_t tot = (size_t)M * C;
    for (size_t i = (size_t)blockIdx.x * blockDim.x + threadIdx.x; i < tot;
         i += (size_t)gridDim.x * blockDim.x) {
        int c = (int)(i % C);
        Z[i] = fmaxf(Z[i] + b[c], 0.0f);
    }
}

// final 256x2 layer + softmax
__global__ void pn_out(const float* __restrict__ h, const float* __restrict__ W,
                       const float* __restrict__ b, float* __restrict__ out) {
    int g = blockIdx.x * blockDim.x + threadIdx.x;
    if (g >= G_GRP) return;
    float l0 = b[0], l1 = b[1];
    for (int c = 0; c < 256; ++c) {
        float v = h[g * 256 + c];
        l0 += v * W[c * 2 + 0];
        l1 += v * W[c * 2 + 1];
    }
    float m  = fmaxf(l0, l1);
    float e0 = expf(l0 - m), e1 = expf(l1 - m);
    float s  = 1.0f / (e0 + e1);
    out[g * 2 + 0] = e0 * s;
    out[g * 2 + 1] = e1 * s;
}

// ---------------------------------------------------------------------------
// Host orchestration
// ---------------------------------------------------------------------------
extern "C" void kernel_launch(void* const* d_in, const int* in_sizes, int n_in,
                              void* d_out, int out_size, void* d_ws, size_t ws_size,
                              hipStream_t stream) {
    (void)in_sizes; (void)n_in; (void)out_size; (void)ws_size;

    const float* x    = (const float*)d_in[0];
    const float* in_g = (const float*)d_in[1];
    const float* in_b = (const float*)d_in[2];
    const float* fcW  = (const float*)d_in[39];
    const float* fcb  = (const float*)d_in[40];
    const float* outW = (const float*)d_in[41];
    const float* outb = (const float*)d_in[42];

    // ---- workspace layout ----
    auto up = [](size_t v) { return (v + 255) & ~(size_t)255; };
    char* base = (char*)d_ws;
    size_t off = 0;
    float*     statsA = (float*)(base + off);      off = up(off + 512 * 4);
    float*     statsS = (float*)(base + off);      off = up(off + 512 * 4);
    float*     fts32  = (float*)(base + off);      off = up(off + (size_t)8192 * 256 * 4);
    _Float16*  ftsh   = (_Float16*)(base + off);   off = up(off + (size_t)8192 * 256 * 2);
    int*       nbr    = (int*)(base + off);        off = up(off + (size_t)131072 * 4);
    _Float16*  WT     = (_Float16*)(base + off);   off = up(off + (size_t)512 * 256 * 2);
    float*     Zbuf   = (float*)(base + off);      off = up(off + (size_t)131072 * 256 * 4);
    _Float16*  Ha     = (_Float16*)(base + off);   off = up(off + (size_t)131072 * 256 * 2);
    _Float16*  Hb     = (_Float16*)(base + off);   off = up(off + (size_t)131072 * 256 * 2);
    float*     Zs     = (float*)(base + off);      off = up(off + (size_t)8192 * 256 * 4);
    _Float16*  pooled = (_Float16*)(base + off);   off = up(off + (size_t)128 * 256 * 2);
    float*     Zfc    = (float*)(base + off);      off = up(off + (size_t)128 * 256 * 4);

    auto ewg = [](size_t tot) {
        size_t g = (tot + 255) / 256;
        return (int)(g > 16384 ? 16384 : (g ? g : 1));
    };
    const int ME = 131072;   // G*N*K edge rows
    const int MS = 8192;     // G*N rows

    // ---- input BN (batch stats over all G*N rows, per feature) ----
    pn_zero<<<2, 256, 0, stream>>>(statsA, 512);
    pn_colstats<<<64, 256, 0, stream>>>(x, MS, 7, statsA, statsA + 256);
    // ftsh for block 1 is stored with K-padded stride 32: pre-zero it
    pn_zero<<<256, 256, 0, stream>>>((float*)ftsh, MS * 32 / 2);
    pn_in_norm<<<ewg((size_t)MS * 7), 256, 0, stream>>>(x, statsA, statsA + 256,
                                                        in_g, in_b, fts32, ftsh,
                                                        MS, 7, 32);

    // ---- three EdgeConv blocks ----
    const int CH[3][3] = {{64, 64, 64}, {128, 128, 128}, {256, 256, 256}};
    int in_c = 7;
    for (int blk = 0; blk < 3; ++blk) {
        const int c0 = CH[blk][0], c1 = CH[blk][1], c2 = CH[blk][2];
        const float* W1 = (const float*)d_in[3 + blk * 12 + 0];
        const float* g1 = (const float*)d_in[3 + blk * 12 + 1];
        const float* b1 = (const float*)d_in[3 + blk * 12 + 2];
        const float* W2 = (const float*)d_in[3 + blk * 12 + 3];
        const float* g2 = (const float*)d_in[3 + blk * 12 + 4];
        const float* b2 = (const float*)d_in[3 + blk * 12 + 5];
        const float* W3 = (const float*)d_in[3 + blk * 12 + 6];
        const float* g3 = (const float*)d_in[3 + blk * 12 + 7];
        const float* b3 = (const float*)d_in[3 + blk * 12 + 8];
        const float* Ws = (const float*)d_in[3 + blk * 12 + 9];
        const float* gs = (const float*)d_in[3 + blk * 12 + 10];
        const float* bs = (const float*)d_in[3 + blk * 12 + 11];

        const int KpS = (in_c % 32) ? 32 : in_c;              // skip-GEMM padded K
        const int Kd  = 2 * in_c;
        const int KpE = (Kd % 32) ? ((Kd + 31) / 32) * 32 : Kd; // edge-GEMM padded K

        // kNN (block 0: raw xyz columns of x; else: current features)
        if (blk == 0)
            pn_knn<<<G_GRP, N_PTS, N_PTS * 3 * 4, stream>>>(x, 7, 3, nbr);
        else
            pn_knn<<<G_GRP, N_PTS, (size_t)N_PTS * in_c * 4, stream>>>(fts32, in_c, in_c, nbr);

        // skip path: Zs = fts @ Ws, stats into statsS (consumed in combine)
        pn_cvt_wT<<<ewg((size_t)c2 * KpS), 256, 0, stream>>>(Ws, WT, in_c, c2, KpS);
        pn_zero<<<2, 256, 0, stream>>>(statsS, 512);
        pn_gemm<<<dim3(MS / 128, c2 / 16), 256, 0, stream>>>(ftsh, WT, Zs, c2, KpS,
                                                             statsS, statsS + 256);

        // layer 1 (fused edge gather)
        pn_cvt_wT<<<ewg((size_t)c0 * KpE), 256, 0, stream>>>(W1, WT, Kd, c0, KpE);
        pn_zero<<<2, 256, 0, stream>>>(statsA, 512);
        pn_gemm_edge<<<dim3(ME / 128, c0 / 16), 256, 0, stream>>>(fts32, nbr, in_c, WT,
                                                                  Zbuf, c0, KpE,
                                                                  statsA, statsA + 256);
        pn_bn_relu<<<ewg((size_t)ME * c0), 256, 0, stream>>>(Zbuf, Ha, statsA, statsA + 256,
                                                             g1, b1, ME, c0);
        // layer 2
        pn_cvt_wT<<<ewg((size_t)c1 * c0), 256, 0, stream>>>(W2, WT, c0, c1, c0);
        pn_zero<<<2, 256, 0, stream>>>(statsA, 512);
        pn_gemm<<<dim3(ME / 128, c1 / 16), 256, 0, stream>>>(Ha, WT, Zbuf, c1, c0,
                                                             statsA, statsA + 256);
        pn_bn_relu<<<ewg((size_t)ME * c1), 256, 0, stream>>>(Zbuf, Hb, statsA, statsA + 256,
                                                             g2, b2, ME, c1);
        // layer 3
        pn_cvt_wT<<<ewg((size_t)c2 * c1), 256, 0, stream>>>(W3, WT, c1, c2, c1);
        pn_zero<<<2, 256, 0, stream>>>(statsA, 512);
        pn_gemm<<<dim3(ME / 128, c2 / 16), 256, 0, stream>>>(Hb, WT, Zbuf, c2, c1,
                                                             statsA, statsA + 256);
        pn_bn_relu<<<ewg((size_t)ME * c2), 256, 0, stream>>>(Zbuf, Ha, statsA, statsA + 256,
                                                             g3, b3, ME, c2);

        // max over K + BN(skip) + ReLU -> new fts
        pn_combine<<<ewg((size_t)MS * c2), 256, 0, stream>>>(Ha, Zs, statsS, statsS + 256,
                                                             gs, bs, fts32, ftsh, MS, c2);
        in_c = c2;
    }

    // ---- head: mean pool, FC + ReLU, 2-class softmax ----
    pn_zero<<<64, 256, 0, stream>>>((float*)pooled, 128 * 256 / 2);  // pad rows 64..127
    pn_pool<<<ewg((size_t)G_GRP * 256), 256, 0, stream>>>(fts32, pooled, 256);
    pn_cvt_wT<<<ewg((size_t)256 * 256), 256, 0, stream>>>(fcW, WT, 256, 256, 256);
    pn_gemm<<<dim3(1, 16), 256, 0, stream>>>(pooled, WT, Zfc, 256, 256,
                                             nullptr, nullptr);
    pn_bias_relu<<<ewg((size_t)G_GRP * 256), 256, 0, stream>>>(Zfc, fcb, G_GRP, 256);
    pn_out<<<1, 64, 0, stream>>>(Zfc, outW, outb, (float*)d_out);
}
